// SPD_CNN_18107582120125
// MI455X (gfx1250) — matrix operand order repaired and checked by reading, not compiled, not run on hardware
//
#include <hip/hip_runtime.h>
#include <hip/hip_bf16.h>
#include <stdint.h>

typedef _Float16 half_t;
typedef __attribute__((ext_vector_type(16))) _Float16 v16h;
typedef __attribute__((ext_vector_type(8)))  _Float16 v8h;
typedef __attribute__((ext_vector_type(8)))  float    v8f;

#define RES    31
#define NEL    (RES*RES)        // 961 elements
#define BATCH  16
#define MROWS  (BATCH*NEL)      // 15376 GEMM rows
#define MP     15424            // MROWS rounded up to 64
#define NDOF   2048             // 2*(RES+1)^2
#define KACC_BYTES ((size_t)BATCH*NDOF*NDOF*sizeof(float))

// CDNA5 async global->LDS copy path (ASYNCcnt). Set to 0 to fall back to the
// synchronous VGPR-staged path.
#define USE_ASYNC_LDS 1

static __device__ __forceinline__ v16h cat16(v8h lo, v8h hi) {
  return __builtin_shufflevector(lo, hi, 0,1,2,3,4,5,6,7,8,9,10,11,12,13,14,15);
}

#if USE_ASYNC_LDS
// One instruction per wave: each lane moves 16B global -> LDS (no VGPR round
// trip, tracked by ASYNCcnt).  LDS address = low 32 bits of the flat address
// (ISA 10.2: LDS aperture uses addr[31:0]).
static __device__ __forceinline__ void async_cp16(const half_t* g, half_t* l) {
  asm volatile("global_load_async_to_lds_b128 %0, %1, off"
               :: "v"((uint32_t)(uintptr_t)l), "v"((uint64_t)(uintptr_t)g)
               : "memory");
}
static __device__ __forceinline__ void wait_async_le4() {
  asm volatile("s_wait_asynccnt 0x4" ::: "memory");
}
static __device__ __forceinline__ void wait_async_0() {
  asm volatile("s_wait_asynccnt 0x0" ::: "memory");
}
#endif

// ---------------------------------------------------------------------------
// Generic 64x64-tile WMMA GEMM: C[M x N](f16) = act(A[M x K] * B[K x N] + bias)
// A is f16 row-major [lda mult 32]; B supplied TRANSPOSED: BT[N x K] row-major.
// All dims padded (M=MP, N mult 64, K mult 32) -> no bounds checks on tiles.
// Double-buffered K-slabs: async fills of slab i+1 overlap WMMA on slab i.
// ---------------------------------------------------------------------------
__global__ __launch_bounds__(128) void gemm16(
    const half_t* __restrict__ A, int lda,
    const half_t* __restrict__ BT, int ldbt,
    const float* __restrict__ bias,
    half_t* __restrict__ Co, int ldc,
    int Nreal, int Ksz, int relu)
{
  __shared__ __align__(16) half_t As[2][64*32];
  __shared__ __align__(16) half_t Bs[2][64*32];   // transposed tile: [n][k]
  const int tid  = threadIdx.x;
  const int wid  = tid >> 5;
  const int lane = tid & 31;
  const int lrow = lane & 15;
  const int hi   = lane >> 4;
  const int row0 = blockIdx.y * 64;
  const int col0 = blockIdx.x * 64;

  // this thread's two 16B chunks of each 64x32 tile (256 chunks / 128 thr)
  const int rr0 = tid >> 2,          ch0 = (tid & 3) * 8;
  const int rr1 = (tid + 128) >> 2,  ch1 = ((tid + 128) & 3) * 8;

  v8f acc[4] = {{}, {}, {}, {}};

#if USE_ASYNC_LDS
  // 4 async instructions per wave per slab fill
  #define GEMM_ISSUE(buf, kk)                                                  \
    do {                                                                       \
      async_cp16(&A [(size_t)(row0 + rr0)*lda  + (kk) + ch0], &As[buf][rr0*32 + ch0]); \
      async_cp16(&A [(size_t)(row0 + rr1)*lda  + (kk) + ch1], &As[buf][rr1*32 + ch1]); \
      async_cp16(&BT[(size_t)(col0 + rr0)*ldbt + (kk) + ch0], &Bs[buf][rr0*32 + ch0]); \
      async_cp16(&BT[(size_t)(col0 + rr1)*ldbt + (kk) + ch1], &Bs[buf][rr1*32 + ch1]); \
    } while (0)

  GEMM_ISSUE(0, 0);
  int buf = 0;
  for (int k0 = 0; k0 < Ksz; k0 += 32) {
    if (k0 + 32 < Ksz) { GEMM_ISSUE(buf ^ 1, k0 + 32); wait_async_le4(); }
    else               { wait_async_0(); }
    __syncthreads();

    const half_t* ap = &As[buf][(wid*16 + lrow)*32 + hi*8];
    v16h af = cat16(*(const v8h*)ap, *(const v8h*)(ap + 16));
#pragma unroll
    for (int t = 0; t < 4; ++t) {
      const half_t* bp = &Bs[buf][(t*16 + lrow)*32 + hi*8];
      v16h bf = cat16(*(const v8h*)bp, *(const v8h*)(bp + 16));
      acc[t] = __builtin_amdgcn_wmma_f32_16x16x32_f16(
          false, af, false, bf, (short)0, acc[t], false, false);
    }
    __syncthreads();      // all waves done reading buf before it is refilled
    buf ^= 1;
  }
  #undef GEMM_ISSUE
#else
  for (int k0 = 0; k0 < Ksz; k0 += 32) {
    uint4 a0 = *(const uint4*)&A [(size_t)(row0 + rr0)*lda  + k0 + ch0];
    uint4 a1 = *(const uint4*)&A [(size_t)(row0 + rr1)*lda  + k0 + ch1];
    uint4 b0 = *(const uint4*)&BT[(size_t)(col0 + rr0)*ldbt + k0 + ch0];
    uint4 b1 = *(const uint4*)&BT[(size_t)(col0 + rr1)*ldbt + k0 + ch1];
    *(uint4*)&As[0][rr0*32 + ch0] = a0;
    *(uint4*)&As[0][rr1*32 + ch1] = a1;
    *(uint4*)&Bs[0][rr0*32 + ch0] = b0;
    *(uint4*)&Bs[0][rr1*32 + ch1] = b1;
    __syncthreads();
    const half_t* ap = &As[0][(wid*16 + lrow)*32 + hi*8];
    v16h af = cat16(*(const v8h*)ap, *(const v8h*)(ap + 16));
#pragma unroll
    for (int t = 0; t < 4; ++t) {
      const half_t* bp = &Bs[0][(t*16 + lrow)*32 + hi*8];
      v16h bf = cat16(*(const v8h*)bp, *(const v8h*)(bp + 16));
      acc[t] = __builtin_amdgcn_wmma_f32_16x16x32_f16(
          false, af, false, bf, (short)0, acc[t], false, false);
    }
    __syncthreads();
  }
#endif

  const int rbase = row0 + wid*16 + hi*8;
#pragma unroll
  for (int t = 0; t < 4; ++t) {
    int   col = col0 + t*16 + lrow;
    float bv  = (bias != nullptr && col < Nreal) ? bias[col] : 0.f;
#pragma unroll
    for (int r = 0; r < 8; ++r) {
      float v = acc[t][r] + bv;
      if (relu) v = fmaxf(v, 0.f);
      Co[(size_t)(rbase + r)*ldc + col] = (half_t)v;
    }
  }
}

// ---------------------------------------------------------------------------
// Weight conversion: f32 -> f16, zero-padded, optional transpose.
// ---------------------------------------------------------------------------
__global__ void cvt_bt(half_t* __restrict__ dst, int ldd, int rtot,
                       const float* __restrict__ src, int srows, int scols,
                       int sld, int trans)
{
  int idx = blockIdx.x * 256 + threadIdx.x;
  if (idx >= rtot * ldd) return;
  int r = idx / ldd, c = idx % ldd;
  float v = 0.f;
  if (r < srows && c < scols)
    v = trans ? src[(size_t)c * sld + r] : src[(size_t)r * sld + c];
  dst[idx] = (half_t)v;
}

// ---------------------------------------------------------------------------
// First conv (k x k, pad 'same', 2 in-ch, WK out-ch) + ReLU -> f16 channel-last
// ---------------------------------------------------------------------------
template<int KK>
__global__ void conv1_k(const float* __restrict__ x, const float* __restrict__ w,
                        const float* __restrict__ bias, half_t* __restrict__ h1)
{
  constexpr int WK  = 16 << ((KK - 1) / 2);
  constexpr int LDH = (WK + 31) / 32 * 32;
  constexpr int PAD = (KK - 1) / 2;
  int idx = blockIdx.x * 256 + threadIdx.x;
  int m = idx / LDH, o = idx % LDH;
  if (m >= MROWS) return;
  if (o >= WK) { h1[(size_t)m*LDH + o] = (half_t)0.f; return; }
  int b = m / NEL, p = m % NEL;
  int pr = p / RES, pc = p % RES;
  const float* x0 = x + (size_t)b * 2 * NEL;
  const float* wo = w + (size_t)o * 2 * KK * KK;
  float acc = bias[o];
#pragma unroll
  for (int kr = 0; kr < KK; ++kr)
#pragma unroll
    for (int kc = 0; kc < KK; ++kc) {
      int ir = pr + kr - PAD, ic = pc + kc - PAD;
      if ((unsigned)ir < (unsigned)RES && (unsigned)ic < (unsigned)RES) {
        acc = fmaf(x0[ir*RES + ic],        wo[kr*KK + kc],           acc);
        acc = fmaf(x0[NEL + ir*RES + ic],  wo[KK*KK + kr*KK + kc],   acc);
      }
    }
  h1[(size_t)m*LDH + o] = (half_t)fmaxf(acc, 0.f);
}

// ---------------------------------------------------------------------------
// Fused SPD + assembly: one block per (batch, element).
//  - async-copy this element's B_flat row into LDS staging
//  - unpack tril into LDS Bm (ReLU diagonal, zero pad)
//  - G = Bm * Bm^T via WMMA, ONLY upper-triangular tile pairs (G symmetric);
//    off-diagonal tiles are scattered twice (mirrored)
//  - scatter-add through loc-map + inner-slice into Kacc with f32 atomics
// ---------------------------------------------------------------------------
template<int KK>
__global__ __launch_bounds__(128) void spd_assemble(
    const half_t* __restrict__ Bflat, int ldb, float* __restrict__ Kacc)
{
  constexpr int OUT  = 2 * (KK + 1) * (KK + 1);      // 8 / 32 / 72
  constexpr int T    = (OUT + 15) / 16;              // tiles per dim
  constexpr int TT   = T * (T + 1) / 2;              // upper-tri tile pairs
  constexpr int OUTP = T * 16;                       // 16 / 32 / 80
  constexpr int KP   = ((OUT + 31) / 32) * 32;       // 32 / 32 / 96
  constexpr int L    = OUT * (OUT + 1) / 2;
  constexpr int LPC  = (L + 63) / 64 * 64;           // padded row length
  constexpr int PAD  = (KK - 1) / 2;
  constexpr int NPN  = KK + 1;

  __shared__ __align__(16) half_t Bm[OUTP * KP];
  __shared__ __align__(16) half_t Lst[LPC];
  const int tid = threadIdx.x;
  const int m   = blockIdx.x;                        // b*961 + e
  const int b   = m / NEL, e = m % NEL;
  const int er  = e / RES, ec = e % RES;

  const half_t* rowp = Bflat + (size_t)m * ldb;
#if USE_ASYNC_LDS
  for (int c = tid; c < LPC / 8; c += 128)           // 16B per lane
    async_cp16(rowp + c * 8, &Lst[c * 8]);
#else
  for (int c = tid; c < LPC / 8; c += 128)
    *(uint4*)&Lst[c * 8] = *(const uint4*)(rowp + c * 8);
#endif

  for (int i = tid; i < OUTP * KP / 8; i += 128) {
    uint4 z; z.x = z.y = z.z = z.w = 0u;
    ((uint4*)Bm)[i] = z;
  }
#if USE_ASYNC_LDS
  wait_async_0();
#endif
  __syncthreads();

  for (int l = tid; l < L; l += 128) {
    float v = (float)Lst[l];
    int i = (int)((sqrtf(8.f * (float)l + 1.f) - 1.f) * 0.5f);
    while ((i + 1) * (i + 2) / 2 <= l) ++i;
    while (i * (i + 1) / 2 > l) --i;
    int j = l - i * (i + 1) / 2;
    if (i == j) v = fmaxf(v, 0.f);                   // PSD factor: relu diag
    Bm[i * KP + j] = (half_t)v;
  }
  __syncthreads();

  const int wid = tid >> 5, lane = tid & 31;
  const int lrow = lane & 15, hi = lane >> 4;

  for (int tp = wid; tp < TT; tp += 4) {
    // decode upper-triangular pair index -> (ti, tj), ti <= tj
    int ti = 0, rem = tp;
    while (rem >= T - ti) { rem -= T - ti; ++ti; }
    const int tj = ti + rem;

    v8f acc = {};
#pragma unroll
    for (int ks = 0; ks < KP; ks += 32) {
      const half_t* ap = &Bm[(ti*16 + lrow)*KP + ks + hi*8];
      v16h af = cat16(*(const v8h*)ap, *(const v8h*)(ap + 16));
      const half_t* bp = &Bm[(tj*16 + lrow)*KP + ks + hi*8];
      v16h bf = cat16(*(const v8h*)bp, *(const v8h*)(bp + 16));
      acc = __builtin_amdgcn_wmma_f32_16x16x32_f16(
          false, af, false, bf, (short)0, acc, false, false);
    }
    // column DOF (fixed per lane for this tile)
    const int jL = tj*16 + lrow;
    int gj = -1;
    if (jL < OUT) {
      int node = jL >> 1, comp = jL & 1;
      int ir2 = er + node / NPN - PAD, ic2 = ec + node % NPN - PAD;
      if ((unsigned)ir2 < 32u && (unsigned)ic2 < 32u)
        gj = ((ir2 * 32 + ic2) << 1) | comp;
    }
#pragma unroll
    for (int r = 0; r < 8; ++r) {
      int iL = ti*16 + r + hi*8;
      if (iL >= OUT || gj < 0) continue;
      int node = iL >> 1, comp = iL & 1;
      int ir2 = er + node / NPN - PAD, ic2 = ec + node % NPN - PAD;
      if ((unsigned)ir2 >= 32u || (unsigned)ic2 >= 32u) continue;   // sliced away
      int gi = ((ir2 * 32 + ic2) << 1) | comp;
      atomicAdd(&Kacc[((size_t)b * NDOF + gi) * NDOF + gj], acc[r]);
      if (ti != tj)   // mirrored entry (G symmetric, strictly off-diagonal)
        atomicAdd(&Kacc[((size_t)b * NDOF + gj) * NDOF + gi], acc[r]);
    }
  }
}

// ---------------------------------------------------------------------------
// Final: out = zero_map ? 0 : Kacc   (float4 / uchar4 streaming)
// ---------------------------------------------------------------------------
__global__ void mask_out_k(const float* __restrict__ Kacc,
                           const unsigned char* __restrict__ zm,
                           float* __restrict__ out)
{
  size_t i = (size_t)blockIdx.x * 256 + threadIdx.x;
  float4 v = ((const float4*)Kacc)[i];
  uchar4 z = ((const uchar4*)zm)[i];
  float4 o;
  o.x = z.x ? 0.f : v.x;
  o.y = z.y ? 0.f : v.y;
  o.z = z.z ? 0.f : v.z;
  o.w = z.w ? 0.f : v.w;
  ((float4*)out)[i] = o;
}

// ---------------------------------------------------------------------------
// Host-side pipeline for one kernel size
// ---------------------------------------------------------------------------
template<int KK>
static void run_k(const float* x, void* const* d_in, int slot,
                  half_t* h1, half_t* h2, half_t* cb, half_t* Bf,
                  half_t* w2t, half_t* w3t, half_t* spdt,
                  float* Kacc, hipStream_t stream)
{
  constexpr int WK   = 16 << ((KK - 1) / 2);
  constexpr int C    = 2 * WK;
  constexpr int LDH1 = (WK + 31) / 32 * 32;    // K of conv2 (padded)
  constexpr int LDH2 = (C + 63) / 64 * 64;     // N of conv2/conv3 (padded)
  constexpr int OUT  = 2 * (KK + 1) * (KK + 1);
  constexpr int L    = OUT * (OUT + 1) / 2;
  constexpr int LP   = (L + 63) / 64 * 64;     // N of SPD gemm (padded)

  const float* w1 = (const float*)d_in[slot + 0];
  const float* b1 = (const float*)d_in[slot + 1];
  const float* w2 = (const float*)d_in[slot + 2];
  const float* b2 = (const float*)d_in[slot + 3];
  const float* w3 = (const float*)d_in[slot + 4];
  const float* b3 = (const float*)d_in[slot + 5];
  const float* sw = (const float*)d_in[slot + 6];

  // weights -> f16, transposed-B layout, zero padded
  cvt_bt<<<(LDH2*LDH1 + 255)/256, 256, 0, stream>>>(w2t, LDH1, LDH2, w2, C, WK, WK, 0);
  cvt_bt<<<(LDH2*C    + 255)/256, 256, 0, stream>>>(w3t, C,    LDH2, w3, C, C,  C,  0);
  cvt_bt<<<(LP*C      + 255)/256, 256, 0, stream>>>(spdt, C,   LP,   sw, L, C,  L,  1);

  conv1_k<KK><<<(MROWS*LDH1 + 255)/256, 256, 0, stream>>>(x, w1, b1, h1);

  // h2 = relu(h1 * W2 + b2)   [15376 x WK] x [WK x C]
  gemm16<<<dim3(LDH2/64, MP/64), 128, 0, stream>>>(h1, LDH1, w2t, LDH1, b2, h2, LDH2, C, LDH1, 1);
  // c  = relu(h2 * W3 + b3)   [15376 x C] x [C x C]
  gemm16<<<dim3(LDH2/64, MP/64), 128, 0, stream>>>(h2, LDH2, w3t, C,    b3, cb, LDH2, C, C,    1);
  // B_flat = c * spdw         [15376 x C] x [C x L]
  gemm16<<<dim3(LP/64,   MP/64), 128, 0, stream>>>(cb, LDH2, spdt, C, nullptr, Bf, LP, L, C,   0);

  spd_assemble<KK><<<MROWS, 128, 0, stream>>>(Bf, LP, Kacc);
}

extern "C" void kernel_launch(void* const* d_in, const int* in_sizes, int n_in,
                              void* d_out, int out_size, void* d_ws, size_t ws_size,
                              hipStream_t stream)
{
  (void)in_sizes; (void)n_in; (void)out_size; (void)ws_size;
  const float*         x  = (const float*)d_in[0];
  const unsigned char* zm = (const unsigned char*)d_in[1];

  char* ws = (char*)d_ws;
  size_t off = 0;
  float*  Kacc = (float*)(ws + off);  off += KACC_BYTES;               // 256 MB
  half_t* h1   = (half_t*)(ws + off); off += (size_t)MP * 64   * 2;    // max wk pad
  half_t* h2   = (half_t*)(ws + off); off += (size_t)MP * 128  * 2;
  half_t* cb   = (half_t*)(ws + off); off += (size_t)MP * 128  * 2;
  half_t* Bf   = (half_t*)(ws + off); off += (size_t)MP * 2688 * 2;    // max Lp
  half_t* w2t  = (half_t*)(ws + off); off += (size_t)128 * 64  * 2;
  half_t* w3t  = (half_t*)(ws + off); off += (size_t)128 * 128 * 2;
  half_t* spdt = (half_t*)(ws + off); off += (size_t)2688 * 128 * 2;

  hipMemsetAsync(Kacc, 0, KACC_BYTES, stream);   // graph-capturable memset node

  run_k<1>(x, d_in, 2,  h1, h2, cb, Bf, w2t, w3t, spdt, Kacc, stream);
  run_k<3>(x, d_in, 9,  h1, h2, cb, Bf, w2t, w3t, spdt, Kacc, stream);
  run_k<5>(x, d_in, 16, h1, h2, cb, Bf, w2t, w3t, spdt, Kacc, stream);

  mask_out_k<<<(size_t)BATCH*NDOF*NDOF/4/256, 256, 0, stream>>>(Kacc, zm, (float*)d_out);
}
